// LanguageModel_72688026517599
// MI455X (gfx1250) — matrix operand order repaired and checked by reading, not compiled, not run on hardware
//
#include <hip/hip_runtime.h>
#include <hip/hip_bf16.h>
#include <math.h>

// ---------------------------------------------------------------------------
// 16-layer transformer forward for MI455X (gfx1250, wave32, WMMA).
// All GEMMs run through v_wmma_f32_16x16x32_bf16 (fp32 accum, bf16 inputs).
// All GEMM dims are exact multiples of the 128x128x32 tiling -> no guards.
// Double-buffered LDS staging: global b128 loads for tile k+1 overlap the
// WMMA work on tile k; one barrier per K-step.
// fp32->bf16 conversion uses native __bf16 casts so the backend can emit
// hardware v_cvt(_pk)_bf16_f32 instead of manual bit-twiddling.
// ---------------------------------------------------------------------------

#define LYR 16
#define DMODEL 768
#define NHEAD 8
#define DHEAD 96
#define FFN 2048
#define VOCAB 32000
#define BATCH 2
#define SEQ 1024
#define MTOK (BATCH * SEQ)   // 2048 rows

typedef __attribute__((ext_vector_type(16))) __bf16 bf16x16;
typedef __attribute__((ext_vector_type(2)))  __bf16 bf16x2;
typedef __attribute__((ext_vector_type(8)))  float  f32x8;

union FragU { unsigned int u[8]; bf16x16 v; };

__device__ __forceinline__ unsigned int pack2bf(float lo, float hi) {
    bf16x2 p;
    p.x = (__bf16)lo;               // RNE hardware convert
    p.y = (__bf16)hi;
    return __builtin_bit_cast(unsigned int, p);
}
__device__ __forceinline__ unsigned short f2bf(float f) {
    __bf16 b = (__bf16)f;
    return __builtin_bit_cast(unsigned short, b);
}

// branchless tanh-GELU: tanh(u) = (t-1)/(t+1), t=e^{2u}  =>  gelu = x*t/(t+1)
__device__ __forceinline__ float gelu_tanh(float x) {
    float u = 0.7978845608028654f * (x + 0.044715f * x * x * x);
    float t = __expf(2.0f * u);                 // v_exp_f32 path
    return x * t * __builtin_amdgcn_rcpf(t + 1.0f);
}

// ---------------------------------------------------------------------------
// Fused token+position embedding + RMSNorm.  One block per token row.
// ---------------------------------------------------------------------------
__global__ __launch_bounds__(256) void embed_rmsnorm_kernel(
    const int* __restrict__ tok, const float* __restrict__ tok_emb,
    const float* __restrict__ pos_emb, const float* __restrict__ w,
    float* __restrict__ out)
{
    const int n = blockIdx.x;           // 0..MTOK-1
    const int s = n % SEQ;
    const int t = tok[n];
    const int tid = threadIdx.x;        // 256 threads, 3 elems each (D=768)

    float e[3];
    float ss = 0.f;
    #pragma unroll
    for (int i = 0; i < 3; ++i) {
        int d = tid + i * 256;
        float v = tok_emb[(size_t)t * DMODEL + d] + pos_emb[(size_t)s * DMODEL + d];
        e[i] = v;
        ss += v * v;
    }
    __shared__ float red[256];
    red[tid] = ss;
    __syncthreads();
    for (int o = 128; o > 0; o >>= 1) {
        if (tid < o) red[tid] += red[tid + o];
        __syncthreads();
    }
    float inv = __builtin_amdgcn_rsqf(red[0] / (float)DMODEL + 1e-6f);
    #pragma unroll
    for (int i = 0; i < 3; ++i) {
        int d = tid + i * 256;
        out[(size_t)n * DMODEL + d] = e[i] * inv * w[d];
    }
}

// ---------------------------------------------------------------------------
// RMSNorm: one block per row, D=768.
// ---------------------------------------------------------------------------
__global__ __launch_bounds__(256) void rmsnorm_kernel(
    const float* __restrict__ x, const float* __restrict__ w,
    float* __restrict__ out)
{
    const int n = blockIdx.x;
    const int tid = threadIdx.x;
    float e[3];
    float ss = 0.f;
    #pragma unroll
    for (int i = 0; i < 3; ++i) {
        int d = tid + i * 256;
        float v = x[(size_t)n * DMODEL + d];
        e[i] = v;
        ss += v * v;
    }
    __shared__ float red[256];
    red[tid] = ss;
    __syncthreads();
    for (int o = 128; o > 0; o >>= 1) {
        if (tid < o) red[tid] += red[tid + o];
        __syncthreads();
    }
    float inv = __builtin_amdgcn_rsqf(red[0] / (float)DMODEL + 1e-6f);
    #pragma unroll
    for (int i = 0; i < 3; ++i) {
        int d = tid + i * 256;
        out[(size_t)n * DMODEL + d] = e[i] * inv * w[d];
    }
}

// ---------------------------------------------------------------------------
// WMMA bf16 GEMM:  C[M,N] = epi( A[M,K] @ B[K,N] (+ Res) )
// Block tile 128x128, K-step 32, 8 waves; each wave = 2x4 grid of 16x16 tiles.
// Requires: M % 128 == 0, N % 128 == 0, K % 32 == 0 (true for this model).
// epi: 0 = none, 1 = add residual, 2 = tanh-GELU.
// ---------------------------------------------------------------------------
#define BM 128
#define BN 128
#define BK 32
#define LDA 36   // ushort stride; 72B rows keep 8-byte-aligned b64 LDS stores
#define LDB 36

__global__ __launch_bounds__(256) void gemm_bf16_wmma(
    const float* __restrict__ A, const float* __restrict__ B,
    const float* __restrict__ Res, float* __restrict__ C,
    int M, int N, int K, int epi)
{
    __shared__ __align__(16) unsigned short As[2][BM * LDA]; // row-major [m][k]
    __shared__ __align__(16) unsigned short Bs[2][BN * LDB]; // col-major [n][k]

    const int tid  = threadIdx.x;
    const int lane = tid & 31;
    const int wid  = tid >> 5;
    const int wr   = wid & 3;   // wave row  (4)  -> 32 rows each
    const int wc   = wid >> 2;  // wave col  (2)  -> 64 cols each
    const int m0   = blockIdx.y * BM;
    const int n0   = blockIdx.x * BN;
    const int lhalf = lane >> 4;   // 0 or 1
    const int l15   = lane & 15;

    // A staging map: 4 float4 per thread; 8 float4 per 32-wide row
    const int a_row = tid >> 3;          // base row (advances by 32 per q)
    const int a_kq  = (tid & 7) * 4;     // k offset within tile (float4)
    // B staging map: 4 float4 per thread; 32 float4 per 128-wide row
    const int b_kk  = tid >> 5;          // base k row (advances by 8 per q)
    const int b_nq  = (tid & 31) * 4;    // n offset within tile (float4)

    f32x8 acc[2][4];
    const f32x8 zero8 = {0.f,0.f,0.f,0.f,0.f,0.f,0.f,0.f};
    #pragma unroll
    for (int i = 0; i < 2; ++i)
        #pragma unroll
        for (int j = 0; j < 4; ++j) acc[i][j] = zero8;

    // ---- prologue: stage tile 0 into buffer 0 ----
    #pragma unroll
    for (int q = 0; q < 4; ++q) {
        int row = a_row + q * 32;
        const float4 va = *reinterpret_cast<const float4*>(
            &A[(size_t)(m0 + row) * K + a_kq]);
        *reinterpret_cast<unsigned long long*>(&As[0][row * LDA + a_kq]) =
            (unsigned long long)pack2bf(va.x, va.y) |
            ((unsigned long long)pack2bf(va.z, va.w) << 32);
    }
    #pragma unroll
    for (int q = 0; q < 4; ++q) {
        int kk = b_kk + q * 8;
        const float4 vb = *reinterpret_cast<const float4*>(
            &B[(size_t)kk * N + n0 + b_nq]);
        Bs[0][(b_nq + 0) * LDB + kk] = f2bf(vb.x);
        Bs[0][(b_nq + 1) * LDB + kk] = f2bf(vb.y);
        Bs[0][(b_nq + 2) * LDB + kk] = f2bf(vb.z);
        Bs[0][(b_nq + 3) * LDB + kk] = f2bf(vb.w);
    }

    int p = 0;
    for (int k0 = 0; k0 < K; k0 += BK) {
        __syncthreads();                       // tile k0 ready in buffer p
        const int  kn      = k0 + BK;
        const bool hasnext = kn < K;

        // ---- issue next tile's global loads early (latency hidden by WMMA) --
        float4 na[4], nb[4];
        if (hasnext) {
            #pragma unroll
            for (int q = 0; q < 4; ++q) {
                int row = a_row + q * 32;
                na[q] = *reinterpret_cast<const float4*>(
                    &A[(size_t)(m0 + row) * K + kn + a_kq]);
            }
            #pragma unroll
            for (int q = 0; q < 4; ++q) {
                int kk = b_kk + q * 8;
                nb[q] = *reinterpret_cast<const float4*>(
                    &B[(size_t)(kn + kk) * N + n0 + b_nq]);
            }
            const int kpre = (kn + BK < K) ? (kn + BK) : kn;
            __builtin_prefetch(&A[(size_t)(m0 + a_row) * K + kpre + a_kq], 0, 0);
            __builtin_prefetch(&B[(size_t)(kpre + b_kk) * N + n0 + b_nq], 0, 0);
        }

        // ---- fragments from buffer p (ISA 7.12.2 layouts) ----
        // A 16x32 bf16: lanes 0-15 = row M=lane, K 0-7 & 16-23; lanes 16-31: +8
        FragU afr[2];
        #pragma unroll
        for (int t = 0; t < 2; ++t) {
            int mrow = wr * 32 + t * 16 + l15;
            #pragma unroll
            for (int r = 0; r < 8; ++r) {
                int kk = ((r & 3) * 2) + ((r >> 2) * 16) + (lhalf * 8);
                afr[t].u[r] = *(const unsigned int*)&As[p][mrow * LDA + kk];
            }
        }
        // B 32x16 bf16: lane holds col N=lane%16; K 0-7 & 16-23 (lanes 0-15) / +8
        FragU bfr[4];
        #pragma unroll
        for (int t = 0; t < 4; ++t) {
            int ncol = wc * 64 + t * 16 + l15;
            #pragma unroll
            for (int r = 0; r < 8; ++r) {
                int kk = ((r & 3) * 2) + ((r >> 2) * 16) + (lhalf * 8);
                bfr[t].u[r] = *(const unsigned int*)&Bs[p][ncol * LDB + kk];
            }
        }

        // ---- 8 WMMAs per wave per K-step ----
        #pragma unroll
        for (int i = 0; i < 2; ++i)
            #pragma unroll
            for (int j = 0; j < 4; ++j)
                acc[i][j] = __builtin_amdgcn_wmma_f32_16x16x32_bf16(
                    false, afr[i].v, false, bfr[j].v,
                    (short)0, acc[i][j], false, false);

        // ---- store next tile into the other buffer ----
        if (hasnext) {
            const int np = p ^ 1;
            #pragma unroll
            for (int q = 0; q < 4; ++q) {
                int row = a_row + q * 32;
                *reinterpret_cast<unsigned long long*>(&As[np][row * LDA + a_kq]) =
                    (unsigned long long)pack2bf(na[q].x, na[q].y) |
                    ((unsigned long long)pack2bf(na[q].z, na[q].w) << 32);
            }
            #pragma unroll
            for (int q = 0; q < 4; ++q) {
                int kk = b_kk + q * 8;
                Bs[np][(b_nq + 0) * LDB + kk] = f2bf(nb[q].x);
                Bs[np][(b_nq + 1) * LDB + kk] = f2bf(nb[q].y);
                Bs[np][(b_nq + 2) * LDB + kk] = f2bf(nb[q].z);
                Bs[np][(b_nq + 3) * LDB + kk] = f2bf(nb[q].w);
            }
        }
        p ^= 1;
    }

    // ---- epilogue (C/D layout: vgpr r -> M=r+8*lanehalf, N=lane&15) ----
    #pragma unroll
    for (int i = 0; i < 2; ++i) {
        #pragma unroll
        for (int j = 0; j < 4; ++j) {
            #pragma unroll
            for (int r = 0; r < 8; ++r) {
                int row = m0 + wr * 32 + i * 16 + r + lhalf * 8;
                int col = n0 + wc * 64 + j * 16 + l15;
                float v = acc[i][j][r];
                if (epi == 1)      v += Res[(size_t)row * N + col];
                else if (epi == 2) v = gelu_tanh(v);
                C[(size_t)row * N + col] = v;
            }
        }
    }
}

// ---------------------------------------------------------------------------
// Causal attention, online softmax. One wave32 per (b,h,q) row.
// q,k,v,o layout: [B,S,H*DH] (GEMM output order). DH=96 -> 3 regs/lane.
// ---------------------------------------------------------------------------
__global__ __launch_bounds__(256) void attn_kernel(
    const float* __restrict__ q, const float* __restrict__ k,
    const float* __restrict__ v, float* __restrict__ o)
{
    const int lane = threadIdx.x & 31;
    const int wid  = threadIdx.x >> 5;
    const int row  = blockIdx.x * 8 + wid;      // 0 .. B*H*S-1
    const int b  = row / (NHEAD * SEQ);
    const int rr = row % (NHEAD * SEQ);
    const int h  = rr / SEQ;
    const int qi = rr % SEQ;
    const float scale = 0.10206207261596575f;   // 96^-0.5

    const size_t rowstride = (size_t)NHEAD * DHEAD;
    const float* qp = q + ((size_t)(b * SEQ + qi)) * rowstride + h * DHEAD;
    const float qv0 = qp[lane], qv1 = qp[lane + 32], qv2 = qp[lane + 64];

    float mrun = -INFINITY, lrun = 0.f;
    float a0 = 0.f, a1 = 0.f, a2 = 0.f;

    for (int j = 0; j <= qi; ++j) {
        const float* kp = k + ((size_t)(b * SEQ + j)) * rowstride + h * DHEAD;
        float d = qv0 * kp[lane] + qv1 * kp[lane + 32] + qv2 * kp[lane + 64];
        #pragma unroll
        for (int off = 16; off > 0; off >>= 1) d += __shfl_xor(d, off, 32);
        float s = d * scale;
        float nm = fmaxf(mrun, s);
        float corr = __expf(mrun - nm);
        float pr   = __expf(s - nm);
        lrun = lrun * corr + pr;
        const float* vp = v + ((size_t)(b * SEQ + j)) * rowstride + h * DHEAD;
        a0 = a0 * corr + pr * vp[lane];
        a1 = a1 * corr + pr * vp[lane + 32];
        a2 = a2 * corr + pr * vp[lane + 64];
        mrun = nm;
    }
    float inv = __builtin_amdgcn_rcpf(lrun);
    float* op = o + ((size_t)(b * SEQ + qi)) * rowstride + h * DHEAD;
    op[lane]      = a0 * inv;
    op[lane + 32] = a1 * inv;
    op[lane + 64] = a2 * inv;
}

// ---------------------------------------------------------------------------
// Host orchestration
// ---------------------------------------------------------------------------
extern "C" void kernel_launch(void* const* d_in, const int* in_sizes, int n_in,
                              void* d_out, int out_size, void* d_ws, size_t ws_size,
                              hipStream_t stream)
{
    (void)in_sizes; (void)n_in; (void)out_size; (void)ws_size;

    const int*   tok        = (const int*)  d_in[0];
    const float* tok_emb    = (const float*)d_in[1];
    const float* pos_emb    = (const float*)d_in[2];
    const float* pos_norm_w = (const float*)d_in[3];
    const float* attn_nw    = (const float*)d_in[4];
    const float* wq         = (const float*)d_in[5];
    const float* wk         = (const float*)d_in[6];
    const float* wv         = (const float*)d_in[7];
    const float* wo         = (const float*)d_in[8];
    const float* ffn_nw     = (const float*)d_in[9];
    const float* w1         = (const float*)d_in[10];
    const float* w2         = (const float*)d_in[11];
    const float* out_nw     = (const float*)d_in[12];
    const float* out_w      = (const float*)d_in[13];
    float*       out        = (float*)d_out;

    const int M = MTOK;
    const size_t MD = (size_t)M * DMODEL * sizeof(float);
    char* p = (char*)d_ws;
    float* xb  = (float*)p; p += MD;                     // residual stream
    float* hb  = (float*)p; p += MD;                     // post-attn residual
    float* xn  = (float*)p; p += MD;                     // normed activations
    float* qb  = (float*)p; p += MD;
    float* kb  = (float*)p; p += MD;
    float* vb  = (float*)p; p += MD;
    float* ob  = (float*)p; p += MD;
    float* mb  = (float*)p; p += (size_t)M * FFN * sizeof(float); // FFN mid

    const dim3 blk(256);
    const dim3 gD(DMODEL / 128, M / 128);   // (6,16)  N=768
    const dim3 gF(FFN    / 128, M / 128);   // (16,16) N=2048
    const dim3 gV(VOCAB  / 128, M / 128);   // (250,16) N=32000

    // embed + first norm
    embed_rmsnorm_kernel<<<M, blk, 0, stream>>>(tok, tok_emb, pos_emb, pos_norm_w, xb);

    for (int l = 0; l < LYR; ++l) {
        const size_t wOff  = (size_t)l * DMODEL * DMODEL;
        const size_t w1Off = (size_t)l * DMODEL * FFN;
        const size_t w2Off = (size_t)l * FFN * DMODEL;

        // attention sub-block
        rmsnorm_kernel<<<M, blk, 0, stream>>>(xb, attn_nw + (size_t)l * DMODEL, xn);
        gemm_bf16_wmma<<<gD, blk, 0, stream>>>(xn, wq + wOff, nullptr, qb, M, DMODEL, DMODEL, 0);
        gemm_bf16_wmma<<<gD, blk, 0, stream>>>(xn, wk + wOff, nullptr, kb, M, DMODEL, DMODEL, 0);
        gemm_bf16_wmma<<<gD, blk, 0, stream>>>(xn, wv + wOff, nullptr, vb, M, DMODEL, DMODEL, 0);
        attn_kernel<<<(BATCH * NHEAD * SEQ) / 8, blk, 0, stream>>>(qb, kb, vb, ob);
        gemm_bf16_wmma<<<gD, blk, 0, stream>>>(ob, wo + wOff, xb, hb, M, DMODEL, DMODEL, 1);

        // FFN sub-block
        rmsnorm_kernel<<<M, blk, 0, stream>>>(hb, ffn_nw + (size_t)l * DMODEL, xn);
        gemm_bf16_wmma<<<gF, blk, 0, stream>>>(xn, w1 + w1Off, nullptr, mb, M, FFN, DMODEL, 2);
        gemm_bf16_wmma<<<gD, blk, 0, stream>>>(mb, w2 + w2Off, hb, xb, M, DMODEL, FFN, 1);
    }

    // final norm + vocab projection
    rmsnorm_kernel<<<M, blk, 0, stream>>>(xb, out_nw, xn);
    gemm_bf16_wmma<<<gV, blk, 0, stream>>>(xn, out_w, nullptr, out, M, VOCAB, DMODEL, 0);
}
